// SelfAttention_3066606650042
// MI455X (gfx1250) — matrix operand order repaired and checked by reading, not compiled, not run on hardware
//
#include <hip/hip_runtime.h>

// ---------------------------------------------------------------------------
// Self-attention for MI455X (gfx1250, wave32, WMMA bf16 w/ fp32 accumulate)
//   B=4, S=2048, E=1024, H=16, D=64
// ---------------------------------------------------------------------------

#define S_LEN   2048
#define EMB     1024
#define NHEAD   16
#define HDIM    64
#define SCALE   0.125f              // 1/sqrt(64)
#define LOG2E   1.4426950408889634f

typedef __attribute__((ext_vector_type(16))) __bf16 bf16x16;
typedef __attribute__((ext_vector_type(8)))  float  f32x8;
typedef __attribute__((ext_vector_type(4)))  int    i32x4;

union FragB16 {
  uint4   u4[2];
  bf16x16 v;
};

#define WMMA_BF16(a, b, c) \
  __builtin_amdgcn_wmma_f32_16x16x32_bf16(false, (a), false, (b), (short)0, (c), false, false)

// Async global->LDS DMA (ASYNCcnt) when the toolchain exposes it; otherwise
// fall back to global->VGPR->ds_store with the same double-buffered schedule.
// Probe-derived signature: (as1 i32x4* gsrc, as3 i32x4* ldst, imm offset, imm cpol)
#if defined(__gfx1250__) && \
    __has_builtin(__builtin_amdgcn_global_load_async_to_lds_b128) && \
    __has_builtin(__builtin_amdgcn_s_wait_asynccnt)
#define USE_ASYNC_LDS 1
typedef __attribute__((address_space(1))) i32x4 as1_i32x4;
typedef __attribute__((address_space(3))) i32x4 as3_i32x4;
#define ASYNC_CP_B128(gsrc, ldst, off) \
  __builtin_amdgcn_global_load_async_to_lds_b128((as1_i32x4*)(gsrc), (as3_i32x4*)(ldst), (off), 0)
#define ASYNC_WAIT() __builtin_amdgcn_s_wait_asynccnt(0)
#else
#define USE_ASYNC_LDS 0
#endif

// ---------------------------------------------------------------------------
// Elementwise fp32 -> bf16 conversion (4 elements / thread, packed b64 store)
// ---------------------------------------------------------------------------
__global__ void k_cvt_bf16(const float* __restrict__ in, __bf16* __restrict__ out, int n) {
  int i = (blockIdx.x * 256 + threadIdx.x) * 4;
  if (i + 3 >= n) return;
  float4 v = *(const float4*)(in + i);
  __bf16 r[4] = {(__bf16)v.x, (__bf16)v.y, (__bf16)v.z, (__bf16)v.w};
  *(uint2*)(out + i) = *(const uint2*)r;
}

// ---------------------------------------------------------------------------
// LDS-tiled transpose + fp32->bf16: in[R][C] -> outT[C][R]
// ---------------------------------------------------------------------------
__global__ void k_transpose_cvt(const float* __restrict__ in, __bf16* __restrict__ outT,
                                int R, int C) {
  __shared__ float s[32][33];
  int tx = threadIdx.x & 31, ty = threadIdx.x >> 5;
  int c0 = blockIdx.x * 32, r0 = blockIdx.y * 32;
#pragma unroll
  for (int i = 0; i < 4; ++i)
    s[ty + 8 * i][tx] = in[(size_t)(r0 + ty + 8 * i) * C + c0 + tx];
  __syncthreads();
#pragma unroll
  for (int i = 0; i < 4; ++i)
    outT[(size_t)(c0 + ty + 8 * i) * R + r0 + tx] = (__bf16)s[tx][ty + 8 * i];
}

// ---------------------------------------------------------------------------
// GEMM mainloop: C[128x128] tile = A[128xK] x Bt[128xK]^T, bf16 in, fp32 acc.
// 256 threads = 8 waves (2 M x 4 N); each wave owns 4x2 tiles of 16x16.
// Double-buffered LDS slabs (k-depth 32), one barrier per iteration; next
// slab streamed via async global->LDS DMA while WMMAs consume current slab.
// ---------------------------------------------------------------------------
__device__ __forceinline__ void gemm_mainloop(const __bf16* __restrict__ Abase,
                                              const __bf16* __restrict__ Bbase,
                                              int Kdim, __bf16* sA, __bf16* sB,
                                              f32x8 acc[4][2]) {
  const int tid  = threadIdx.x;
  const int lane = tid & 31;
  const int ln   = lane & 15;
  const int g    = lane >> 4;
  const int wv   = tid >> 5;
  const int wvM  = wv >> 2;   // 0..1
  const int wvN  = wv & 3;    // 0..3

  const int ldRow  = tid >> 1;         // 0..127
  const int ldHalf = (tid & 1) * 16;   // element offset in 32-wide k slab

  const __bf16* aPtr = Abase + (size_t)ldRow * Kdim + ldHalf;
  const __bf16* bPtr = Bbase + (size_t)ldRow * Kdim + ldHalf;
  const int sOff = ldRow * 32 + ldHalf;

  // ---- prologue: stage slab 0 into buffer 0 ----
#if USE_ASYNC_LDS
  ASYNC_CP_B128(aPtr, sA + sOff, 0);
  ASYNC_CP_B128(aPtr, sA + sOff, 16);
  ASYNC_CP_B128(bPtr, sB + sOff, 0);
  ASYNC_CP_B128(bPtr, sB + sOff, 16);
  ASYNC_WAIT();
#else
  {
    uint4 a0 = *(const uint4*)(aPtr);
    uint4 a1 = *(const uint4*)(aPtr + 8);
    uint4 b0 = *(const uint4*)(bPtr);
    uint4 b1 = *(const uint4*)(bPtr + 8);
    *(uint4*)(sA + sOff)     = a0;
    *(uint4*)(sA + sOff + 8) = a1;
    *(uint4*)(sB + sOff)     = b0;
    *(uint4*)(sB + sOff + 8) = b1;
  }
#endif
  __syncthreads();

  int cur = 0;
  for (int k0 = 0; k0 < Kdim; k0 += 32) {
    const int  nxt     = cur ^ 1;
    const bool hasNext = (k0 + 32) < Kdim;
    const __bf16* sAc = sA + cur * (128 * 32);
    const __bf16* sBc = sB + cur * (128 * 32);

    if (k0 + 64 < Kdim) {                       // keep L2 two slabs ahead
      __builtin_prefetch(aPtr + k0 + 64, 0, 3);
      __builtin_prefetch(bPtr + k0 + 64, 0, 3);
    }

#if USE_ASYNC_LDS
    if (hasNext) {                              // DMA next slab, no VGPR staging
      __bf16* la = sA + nxt * (128 * 32) + sOff;
      __bf16* lb = sB + nxt * (128 * 32) + sOff;
      ASYNC_CP_B128(aPtr + k0 + 32, la, 0);
      ASYNC_CP_B128(aPtr + k0 + 32, la, 16);
      ASYNC_CP_B128(bPtr + k0 + 32, lb, 0);
      ASYNC_CP_B128(bPtr + k0 + 32, lb, 16);
    }
#else
    uint4 a0{}, a1{}, b0{}, b1{};
    if (hasNext) {                              // issue loads before compute
      a0 = *(const uint4*)(aPtr + k0 + 32);
      a1 = *(const uint4*)(aPtr + k0 + 40);
      b0 = *(const uint4*)(bPtr + k0 + 32);
      b1 = *(const uint4*)(bPtr + k0 + 40);
    }
#endif

    FragB16 af[4], bf[2];
#pragma unroll
    for (int im = 0; im < 4; ++im) {            // A: lane ln = row, k chunks g*8 / 16+g*8
      int r = wvM * 64 + im * 16 + ln;
      af[im].u4[0] = *(const uint4*)(sAc + r * 32 + g * 8);
      af[im].u4[1] = *(const uint4*)(sAc + r * 32 + 16 + g * 8);
    }
#pragma unroll
    for (int in = 0; in < 2; ++in) {            // B: lane ln = col, k chunks g*8 / 16+g*8
      int r = wvN * 32 + in * 16 + ln;
      bf[in].u4[0] = *(const uint4*)(sBc + r * 32 + g * 8);
      bf[in].u4[1] = *(const uint4*)(sBc + r * 32 + 16 + g * 8);
    }
#pragma unroll
    for (int im = 0; im < 4; ++im)
#pragma unroll
      for (int in = 0; in < 2; ++in)
        acc[im][in] = WMMA_BF16(af[im].v, bf[in].v, acc[im][in]);

    if (hasNext) {
#if USE_ASYNC_LDS
      ASYNC_WAIT();
#else
      __bf16* la = sA + nxt * (128 * 32) + sOff;
      __bf16* lb = sB + nxt * (128 * 32) + sOff;
      *(uint4*)(la)     = a0;
      *(uint4*)(la + 8) = a1;
      *(uint4*)(lb)     = b0;
      *(uint4*)(lb + 8) = b1;
#endif
      __syncthreads();
    }
    cur = nxt;
  }
}

// ---------------------------------------------------------------------------
// Kernel 1: qkv = x @ w_qkv + b_qkv  -> Q/K/V in [B][H][S][D] bf16
// ---------------------------------------------------------------------------
__global__ __launch_bounds__(256, 2)
void k_gemm_qkv(const __bf16* __restrict__ xb, const __bf16* __restrict__ wt,
                const float* __restrict__ bias,
                __bf16* __restrict__ Qb, __bf16* __restrict__ Kb, __bf16* __restrict__ Vb) {
  __shared__ __align__(16) __bf16 sA[2 * 128 * 32];
  __shared__ __align__(16) __bf16 sB[2 * 128 * 32];

  f32x8 acc[4][2];
#pragma unroll
  for (int im = 0; im < 4; ++im)
#pragma unroll
    for (int in = 0; in < 2; ++in)
#pragma unroll
      for (int e = 0; e < 8; ++e) acc[im][in][e] = 0.f;

  gemm_mainloop(xb + (size_t)blockIdx.y * 128 * EMB,
                wt + (size_t)blockIdx.x * 128 * EMB, EMB, sA, sB, acc);

  const int lane = threadIdx.x & 31;
  const int ln = lane & 15, g = lane >> 4;
  const int wv = threadIdx.x >> 5, wvM = wv >> 2, wvN = wv & 3;

#pragma unroll
  for (int im = 0; im < 4; ++im) {
#pragma unroll
    for (int in = 0; in < 2; ++in) {
      int n = blockIdx.x * 128 + wvN * 32 + in * 16 + ln;   // 0..3071
      float bv = bias[n];
      int part = n >> 10;          // 0=Q 1=K 2=V
      int e = n & 1023;
      int h = e >> 6, d = e & 63;
      __bf16* dst = (part == 0) ? Qb : ((part == 1) ? Kb : Vb);
#pragma unroll
      for (int v = 0; v < 8; ++v) {
        int mrow = blockIdx.y * 128 + wvM * 64 + im * 16 + v + 8 * g;
        int b = mrow >> 11, s = mrow & 2047;
        dst[(((size_t)(b * NHEAD + h) * S_LEN + s) << 6) + d] =
            (__bf16)(acc[im][in][v] + bv);
      }
    }
  }
}

// ---------------------------------------------------------------------------
// Kernel 2: flash attention. grid = (S/128, B*H); 8 waves, 16 q-rows each.
// Per 64-key chunk: S = Q K^T (8 wmma), online softmax, O^T += V^T P^T (8 wmma).
// ---------------------------------------------------------------------------
__global__ __launch_bounds__(256, 2)
void k_attn(const __bf16* __restrict__ Qb, const __bf16* __restrict__ Kb,
            const __bf16* __restrict__ Vb, __bf16* __restrict__ attb) {
  __shared__ __align__(16) __bf16 sK[64 * 64];     // [key][d]
  __shared__ __align__(16) __bf16 sVT[64 * 64];    // [d][key]
  __shared__ __align__(16) __bf16 sP[8][16 * 64];  // per wave: [q][key]
  __shared__ float sStat[8][16];                   // per wave per-q scalars

  const int tid = threadIdx.x;
  const int wv = tid >> 5, lane = tid & 31;
  const int ln = lane & 15, g = lane >> 4;

  const int bh = blockIdx.y;               // b*16 + h
  const int b = bh >> 4, h = bh & 15;
  const size_t headOff = (size_t)bh * S_LEN * HDIM;
  const int qbase = blockIdx.x * 128 + wv * 16;

  // Q A-fragments: rows qbase+ln, d split into two 32-deep k-steps
  FragB16 qf[2];
  {
    const __bf16* qrow = Qb + headOff + (size_t)(qbase + ln) * HDIM;
    qf[0].u4[0] = *(const uint4*)(qrow + g * 8);
    qf[0].u4[1] = *(const uint4*)(qrow + 16 + g * 8);
    qf[1].u4[0] = *(const uint4*)(qrow + 32 + g * 8);
    qf[1].u4[1] = *(const uint4*)(qrow + 48 + g * 8);
  }

  f32x8 o[4];                       // O^T tiles: d rows, q cols
#pragma unroll
  for (int t = 0; t < 4; ++t)
#pragma unroll
    for (int e = 0; e < 8; ++e) o[t][e] = 0.f;
  float m[8], l[8];
#pragma unroll
  for (int v = 0; v < 8; ++v) { m[v] = -1e30f; l[v] = 0.f; }
  const float cs = SCALE * LOG2E;   // work in exp2 domain

  const int key = tid >> 2;               // cooperative K/V chunk loading
  const int dpart = (tid & 3) * 16;

  for (int kc = 0; kc < S_LEN / 64; ++kc) {
    __syncthreads();
    {
      const size_t src = headOff + (size_t)(kc * 64 + key) * HDIM + dpart;
#if USE_ASYNC_LDS
      ASYNC_CP_B128(Kb + src, sK + key * 64 + dpart, 0);   // K: straight copy via DMA
      ASYNC_CP_B128(Kb + src, sK + key * 64 + dpart, 16);
#else
      uint4 k0 = *(const uint4*)(Kb + src);
      uint4 k1 = *(const uint4*)(Kb + src + 8);
      *(uint4*)(sK + key * 64 + dpart)     = k0;
      *(uint4*)(sK + key * 64 + dpart + 8) = k1;
#endif
      union { uint4 u[2]; __bf16 e[16]; } vt;              // V: transpose into LDS
      vt.u[0] = *(const uint4*)(Vb + src);
      vt.u[1] = *(const uint4*)(Vb + src + 8);
#pragma unroll
      for (int j = 0; j < 16; ++j) sVT[(dpart + j) * 64 + key] = vt.e[j];
#if USE_ASYNC_LDS
      ASYNC_WAIT();
#endif
    }
    __syncthreads();

    // ---- scores: 4 key-tiles x 2 d-steps ----
    f32x8 s[4];
#pragma unroll
    for (int kt = 0; kt < 4; ++kt)
#pragma unroll
      for (int e = 0; e < 8; ++e) s[kt][e] = 0.f;
#pragma unroll
    for (int ks = 0; ks < 2; ++ks) {
#pragma unroll
      for (int kt = 0; kt < 4; ++kt) {
        FragB16 kb;                               // B = K^T: lane = key col
        int base = (kt * 16 + ln) * 64 + ks * 32 + g * 8;
        kb.u4[0] = *(const uint4*)(sK + base);
        kb.u4[1] = *(const uint4*)(sK + base + 16);
        s[kt] = WMMA_BF16(qf[ks].v, kb.v, s[kt]);
      }
    }
#pragma unroll
    for (int kt = 0; kt < 4; ++kt)
#pragma unroll
      for (int e = 0; e < 8; ++e) s[kt][e] *= cs;

    // ---- online softmax: row stats live at (VGPR v, lane-half g) ----
    float a[8];
#pragma unroll
    for (int v = 0; v < 8; ++v) {
      float mc = s[0][v];
#pragma unroll
      for (int kt = 1; kt < 4; ++kt) mc = fmaxf(mc, s[kt][v]);
#pragma unroll
      for (int off = 1; off < 16; off <<= 1)
        mc = fmaxf(mc, __shfl_xor(mc, off, 16));
      float mn = fmaxf(m[v], mc);
      a[v] = exp2f(m[v] - mn);
      m[v] = mn;
    }
    if (ln == 0) {
#pragma unroll
      for (int v = 0; v < 8; ++v) sStat[wv][g * 8 + v] = a[v];
    }
    float aq = sStat[wv][ln];                 // alpha for this lane's q column
#pragma unroll
    for (int t = 0; t < 4; ++t)
#pragma unroll
      for (int e = 0; e < 8; ++e) o[t][e] *= aq;

#pragma unroll
    for (int v = 0; v < 8; ++v) {
      float rs = 0.f;
#pragma unroll
      for (int kt = 0; kt < 4; ++kt) {
        float p = exp2f(s[kt][v] - m[v]);
        rs += p;
        sP[wv][(v + 8 * g) * 64 + kt * 16 + ln] = (__bf16)p;
      }
#pragma unroll
      for (int off = 1; off < 16; off <<= 1) rs += __shfl_xor(rs, off, 16);
      l[v] = l[v] * a[v] + rs;
    }

    // ---- O^T += V^T x P^T ----
#pragma unroll
    for (int ks = 0; ks < 2; ++ks) {
      FragB16 pb;                               // B = P^T: lane = q col
      int pbase = ln * 64 + ks * 32 + g * 16;
      pb.u4[0] = *(const uint4*)(sP[wv] + pbase);
      pb.u4[1] = *(const uint4*)(sP[wv] + pbase + 8);
#pragma unroll
      for (int dt = 0; dt < 4; ++dt) {
        FragB16 va;                             // A = V^T: lane = d row
        int vbase = (dt * 16 + ln) * 64 + ks * 32;
        va.u4[0] = *(const uint4*)(sVT + vbase + g * 16);
        va.u4[1] = *(const uint4*)(sVT + vbase + g * 16 + 8);
        o[dt] = WMMA_BF16(va.v, pb.v, o[dt]);
      }
    }
  }

  // ---- normalize + store merged-heads bf16 [B][S][E] ----
  if (ln == 0) {
#pragma unroll
    for (int v = 0; v < 8; ++v) sStat[wv][g * 8 + v] = l[v];
  }
  float invl = 1.0f / sStat[wv][ln];
  size_t rowbase = ((size_t)b * S_LEN + (qbase + ln)) * EMB + h * HDIM;
#pragma unroll
  for (int dt = 0; dt < 4; ++dt)
#pragma unroll
    for (int v = 0; v < 8; ++v)
      attb[rowbase + dt * 16 + 8 * g + v] = (__bf16)(o[dt][v] * invl);
}

// ---------------------------------------------------------------------------
// Kernel 3: out = att @ w_out + b_out (fp32 output)
// ---------------------------------------------------------------------------
__global__ __launch_bounds__(256, 2)
void k_gemm_out(const __bf16* __restrict__ attb, const __bf16* __restrict__ wt,
                const float* __restrict__ bias, float* __restrict__ out) {
  __shared__ __align__(16) __bf16 sA[2 * 128 * 32];
  __shared__ __align__(16) __bf16 sB[2 * 128 * 32];

  f32x8 acc[4][2];
#pragma unroll
  for (int im = 0; im < 4; ++im)
#pragma unroll
    for (int in = 0; in < 2; ++in)
#pragma unroll
      for (int e = 0; e < 8; ++e) acc[im][in][e] = 0.f;

  gemm_mainloop(attb + (size_t)blockIdx.y * 128 * EMB,
                wt + (size_t)blockIdx.x * 128 * EMB, EMB, sA, sB, acc);

  const int lane = threadIdx.x & 31;
  const int ln = lane & 15, g = lane >> 4;
  const int wv = threadIdx.x >> 5, wvM = wv >> 2, wvN = wv & 3;

#pragma unroll
  for (int im = 0; im < 4; ++im) {
#pragma unroll
    for (int in = 0; in < 2; ++in) {
      int n = blockIdx.x * 128 + wvN * 32 + in * 16 + ln;
      float bv = bias[n];
#pragma unroll
      for (int v = 0; v < 8; ++v) {
        int mrow = blockIdx.y * 128 + wvM * 64 + im * 16 + v + 8 * g;
        out[(size_t)mrow * EMB + n] = acc[im][in][v] + bv;
      }
    }
  }
}

// ---------------------------------------------------------------------------
extern "C" void kernel_launch(void* const* d_in, const int* in_sizes, int n_in,
                              void* d_out, int out_size, void* d_ws, size_t ws_size,
                              hipStream_t stream) {
  const float* x     = (const float*)d_in[0];   // [4,2048,1024]
  const float* w_qkv = (const float*)d_in[1];   // [1024,3072]
  const float* b_qkv = (const float*)d_in[2];   // [3072]
  const float* w_out = (const float*)d_in[3];   // [1024,1024]
  const float* b_out = (const float*)d_in[4];   // [1024]
  float* out = (float*)d_out;

  char* ws = (char*)d_ws;
  const size_t MiB = 1ull << 20;
  __bf16* xb    = (__bf16*)(ws);                // 16 MiB  [8192][1024]
  __bf16* wqkvT = (__bf16*)(ws + 16 * MiB);     //  6 MiB  [3072][1024]
  __bf16* woutT = (__bf16*)(ws + 22 * MiB);     //  2 MiB  [1024][1024]
  __bf16* Qb    = (__bf16*)(ws + 24 * MiB);     // 16 MiB  [B][H][S][D]
  __bf16* Kb    = (__bf16*)(ws + 40 * MiB);     // 16 MiB
  __bf16* Vb    = (__bf16*)(ws + 56 * MiB);     // 16 MiB
  __bf16* attb  = (__bf16*)(ws + 72 * MiB);     // 16 MiB  [8192][1024]

  const int nX = 4 * S_LEN * EMB;               // 8388608
  k_cvt_bf16<<<nX / (256 * 4), 256, 0, stream>>>(x, xb, nX);
  k_transpose_cvt<<<dim3(3 * EMB / 32, EMB / 32), 256, 0, stream>>>(w_qkv, wqkvT, EMB, 3 * EMB);
  k_transpose_cvt<<<dim3(EMB / 32, EMB / 32), 256, 0, stream>>>(w_out, woutT, EMB, EMB);

  k_gemm_qkv<<<dim3(3 * EMB / 128, 4 * S_LEN / 128), 256, 0, stream>>>(
      xb, wqkvT, b_qkv, Qb, Kb, Vb);

  k_attn<<<dim3(S_LEN / 128, 4 * NHEAD), 256, 0, stream>>>(Qb, Kb, Vb, attb);

  k_gemm_out<<<dim3(EMB / 128, 4 * S_LEN / 128), 256, 0, stream>>>(
      attb, woutT, b_out, out);
}
